// MentionAttention_37787122270992
// MI455X (gfx1250) — matrix operand order repaired and checked by reading, not compiled, not run on hardware
//
#include <hip/hip_runtime.h>
#include <hip/hip_bf16.h>

typedef __attribute__((ext_vector_type(16))) _Float16 v16h;
typedef __attribute__((ext_vector_type(8)))  _Float16 v8h;
typedef __attribute__((ext_vector_type(8)))  float    v8f;

namespace {
constexpr int B = 4, P = 8192, M = 1024, H = 1024, R = 97;
constexpr int PT = P / 16;   // 512 p-row tiles per batch
constexpr int MT = M / 16;   // 64
constexpr int HT = H / 16;   // 64
constexpr int KH = H / 32;   // 32 K-chunks over H
constexpr int KM = M / 32;   // 32 K-chunks over M
constexpr float SCALE = 0.03125f;   // 1024^-0.5
constexpr float MAXC  = 1e10f;
}

// A-fragment (16x32 f16) K index for element e, lane-half h:
// VGPR0..3 -> K = h*8 + e (e<8), VGPR4..7 -> K = 16 + h*8 + (e-8)
__device__ __forceinline__ int kmapA(int e, int half) {
  return ((e & 8) << 1) + half * 8 + (e & 7);
}
// B-fragment (32x16 f16): same K order as A: e<8 -> K = h*8 + e, e>=8 -> K = 16 + h*8 + (e-8)
__device__ __forceinline__ int kmapB(int e, int half) {
  return ((e & 8) << 1) + half * 8 + (e & 7);
}

__device__ __forceinline__ void wait_asynccnt0() {
#if __has_builtin(__builtin_amdgcn_s_wait_asynccnt)
  __builtin_amdgcn_s_wait_asynccnt(0);
#else
  asm volatile("s_wait_asynccnt 0" ::: "memory");
#endif
}

// Async-DMA one lane-slice (32B) of a 1KB fragment chunk into LDS.
// INST_OFFSET is added to BOTH the LDS and global addresses (ISA 10.7.4).
__device__ __forceinline__ void async_copy_frag(uint32_t lds_addr, uint64_t gaddr) {
  asm volatile("global_load_async_to_lds_b128 %0, %1, off"
               :: "v"(lds_addr), "v"(gaddr) : "memory");
  asm volatile("global_load_async_to_lds_b128 %0, %1, off offset:16"
               :: "v"(lds_addr), "v"(gaddr) : "memory");
}

// ---------------------------------------------------------------------------
// Kernel 1: q = res @ q_w + q_b  -> f16, stored in A-fragment layout
//   qf[((b*PT+pt)*KH + h32)*512 + lane*16 + e]
// ---------------------------------------------------------------------------
__global__ __launch_bounds__(128) void qproj_kernel(
    const float* __restrict__ res, const float* __restrict__ q_w,
    const float* __restrict__ q_b, _Float16* __restrict__ qf) {
  const int lane = threadIdx.x & 31, wv = threadIdx.x >> 5;
  const int half = lane >> 4, rl = lane & 15;
  const int gid = blockIdx.x * 4 + wv;      // one 16x16 output tile per wave
  const int n16 = gid % HT;
  const int rowTile = gid / HT;             // [0, B*PT)
  const int b = rowTile / PT, pt = rowTile % PT;
  const size_t arow = ((size_t)b * P + (size_t)pt * 16 + rl) * R;

  v8f acc = {};
  #pragma unroll
  for (int kb = 0; kb < 128; kb += 32) {    // K=97 padded to 128
    v16h a, bb;
    #pragma unroll
    for (int e = 0; e < 16; ++e) {
      const int ka = kb + kmapA(e, half);
      a[e] = (ka < R) ? (_Float16)res[arow + ka] : (_Float16)0.f;
      const int kk = kb + kmapB(e, half);
      bb[e] = (kk < R) ? (_Float16)q_w[(size_t)kk * H + n16 * 16 + rl]
                       : (_Float16)0.f;
    }
    acc = __builtin_amdgcn_wmma_f32_16x16x32_f16(false, a, false, bb,
                                                 (short)0, acc, false, false);
  }
  const float bias = q_b[n16 * 16 + rl];

  // transpose C-fragment (rows in VGPRs) into A-fragment (rows in lanes) via LDS
  __shared__ _Float16 ldsT[4][16][16];
  #pragma unroll
  for (int r = 0; r < 8; ++r) ldsT[wv][r + half * 8][rl] = (_Float16)(acc[r] + bias);
  __builtin_amdgcn_wave_barrier();
  alignas(16) _Float16 ov[8];
  #pragma unroll
  for (int e = 0; e < 8; ++e) ov[e] = ldsT[wv][rl][half * 8 + e];
  const size_t base = ((size_t)(b * PT + pt) * KH + (n16 >> 1)) * 512
                      + lane * 16 + (n16 & 1) * 8;
  *(v8h*)(qf + base) = *(const v8h*)ov;
}

// ---------------------------------------------------------------------------
// Kernel 2: k = mention @ k_w + k_b -> f16, stored in B-fragment layout
//   kf[((b*MT+mtile)*KH + h32)*512 + lane*16 + e]  (lane = m column, K = h)
// ---------------------------------------------------------------------------
__global__ __launch_bounds__(128) void kproj_kernel(
    const float* __restrict__ mention, const float* __restrict__ k_w,
    const float* __restrict__ k_b, _Float16* __restrict__ kf) {
  const int lane = threadIdx.x & 31, wv = threadIdx.x >> 5;
  const int half = lane >> 4, rl = lane & 15;
  const int gid = blockIdx.x * 4 + wv;
  const int n16 = gid % HT;
  const int rowTile = gid / HT;             // [0, B*MT)
  const int b = rowTile / MT, mt = rowTile % MT;
  const size_t arow = ((size_t)b * M + (size_t)mt * 16 + rl) * H;

  v8f acc = {};
  for (int kb = 0; kb < H; kb += 32) {
    v16h a, bb;
    #pragma unroll
    for (int e = 0; e < 16; ++e) {
      a[e]  = (_Float16)mention[arow + kb + kmapA(e, half)];
      bb[e] = (_Float16)k_w[(size_t)(kb + kmapB(e, half)) * H + n16 * 16 + rl];
    }
    acc = __builtin_amdgcn_wmma_f32_16x16x32_f16(false, a, false, bb,
                                                 (short)0, acc, false, false);
  }
  const float bias = k_b[n16 * 16 + rl];

  __shared__ _Float16 ldsT[4][16][16];
  #pragma unroll
  for (int r = 0; r < 8; ++r) ldsT[wv][r + half * 8][rl] = (_Float16)(acc[r] + bias);
  __builtin_amdgcn_wave_barrier();
  alignas(16) _Float16 ov[8];
  #pragma unroll
  for (int e = 0; e < 8; ++e) ov[e] = ldsT[wv][rl][half * 8 + e];
  // B-frag: target lane = half*16 + m_local, elements e = (n16&1)*8 + j
  const size_t base = ((size_t)(b * MT + mt) * KH + (n16 >> 1)) * 512
                      + (half * 16 + rl) * 16 + (n16 & 1) * 8;
  *(v8h*)(kf + base) = *(const v8h*)ov;
}

// ---------------------------------------------------------------------------
// Kernel 3: mention -> f16 B-fragment layout for the PV GEMM
//   mf[((b*HT+ht)*KM + m32)*512 + lane*16 + e]  (lane = h column, K = m)
// ---------------------------------------------------------------------------
__global__ __launch_bounds__(128) void mconv_kernel(
    const float* __restrict__ mention, _Float16* __restrict__ mf) {
  const int lane = threadIdx.x & 31, wv = threadIdx.x >> 5;
  const int half = lane >> 4, rl = lane & 15;
  const int gid = blockIdx.x * 4 + wv;      // [0, B*HT*KM)
  const int m32 = gid % KM;
  const int rem = gid / KM;
  const int ht = rem % HT, b = rem / HT;
  alignas(32) _Float16 vals[16];
  #pragma unroll
  for (int e = 0; e < 16; ++e) {
    const int m = m32 * 32 + kmapB(e, half);
    vals[e] = (_Float16)mention[((size_t)b * M + m) * H + ht * 16 + rl];
  }
  const size_t base = ((size_t)(b * HT + ht) * KM + m32) * 512 + lane * 16;
  *(v16h*)(mf + base) = *(const v16h*)vals;
}

// ---------------------------------------------------------------------------
// Kernel 4: att = q@k^T, masked dual softmax (unnormalized), probs -> f16
// A-fragments; row sums -> rs.  8 waves per block, each owns 128 M-columns.
// q fragments are DMA'd once per block into LDS (double-buffered, ASYNCcnt)
// instead of 8 redundant global loads.
// v=0: mask=h_mask (produces t_att -> output h), v=1: mask=t_mask (-> t)
// ---------------------------------------------------------------------------
__global__ __launch_bounds__(256, 1) void attn_softmax_kernel(
    const _Float16* __restrict__ qf, const _Float16* __restrict__ kf,
    const float* __restrict__ h_mask, const float* __restrict__ t_mask,
    _Float16* __restrict__ pf, float* __restrict__ rs) {
  const int b = blockIdx.x / PT, pt = blockIdx.x % PT;
  const int wv = threadIdx.x >> 5, lane = threadIdx.x & 31;
  const int half = lane >> 4, rl = lane & 15;

  __shared__ _Float16 qlds[2][512];          // double-buffered q A-fragment
  __shared__ float red[8][16];
  __shared__ _Float16 ldsT[8][16][16];

  v8f zero = {};
  v8f S[8];
  #pragma unroll
  for (int t = 0; t < 8; ++t) S[t] = zero;

  const size_t qbase = (size_t)(b * PT + pt) * KH * 512;   // halves
  const uint32_t qlds_a0 = (uint32_t)(uintptr_t)(&qlds[0][lane * 16]);
  const uint32_t qlds_a1 = (uint32_t)(uintptr_t)(&qlds[1][lane * 16]);

  // prologue: wave 0 DMAs chunk 0 into buffer 0
  if (wv == 0) {
    async_copy_frag(qlds_a0, (uint64_t)(uintptr_t)(qf + qbase + lane * 16));
  }

  for (int k32 = 0; k32 < KH; ++k32) {
    const int cur = k32 & 1;
    if (wv == 0) wait_asynccnt0();           // chunk k32 resident in qlds[cur]
    __syncthreads();                          // publish to all 8 waves
    if (wv == 0 && k32 + 1 < KH) {            // overlap next DMA with compute
      async_copy_frag(cur ? qlds_a0 : qlds_a1,
                      (uint64_t)(uintptr_t)(qf + qbase + (size_t)(k32 + 1) * 512
                                            + lane * 16));
    }
    v16h a = *(const v16h*)(&qlds[cur][lane * 16]);
    #pragma unroll
    for (int t = 0; t < 8; ++t) {
      const int mtile = wv * 8 + t;
      v16h bb = *(const v16h*)(kf + ((size_t)(b * MT + mtile) * KH + k32) * 512
                               + lane * 16);
      S[t] = __builtin_amdgcn_wmma_f32_16x16x32_f16(false, a, false, bb,
                                                    (short)0, S[t], false, false);
    }
  }

  for (int v = 0; v < 2; ++v) {
    const float* __restrict__ mask = (v == 0) ? h_mask : t_mask;
    v8f W[8];
    #pragma unroll
    for (int t = 0; t < 8; ++t) {
      const int m = wv * 128 + t * 16 + rl;
      #pragma unroll
      for (int r = 0; r < 8; ++r) {
        const int prow = pt * 16 + r + half * 8;
        const float mv = mask[((size_t)b * P + prow) * M + m];
        W[t][r] = S[t][r] * SCALE - (1.0f - mv) * MAXC;
      }
    }
    // masked row max: local tiles, 16-lane shuffle reduce, cross-wave via LDS
    float rm[8];
    #pragma unroll
    for (int r = 0; r < 8; ++r) {
      float x = W[0][r];
      #pragma unroll
      for (int t = 1; t < 8; ++t) x = fmaxf(x, W[t][r]);
      x = fmaxf(x, __shfl_xor(x, 1, 32));
      x = fmaxf(x, __shfl_xor(x, 2, 32));
      x = fmaxf(x, __shfl_xor(x, 4, 32));
      x = fmaxf(x, __shfl_xor(x, 8, 32));
      rm[r] = x;
    }
    if (rl == 0) {
      #pragma unroll
      for (int r = 0; r < 8; ++r) red[wv][r + half * 8] = rm[r];
    }
    __syncthreads();
    #pragma unroll
    for (int r = 0; r < 8; ++r) {
      float x = red[0][r + half * 8];
      #pragma unroll
      for (int wi = 1; wi < 8; ++wi) x = fmaxf(x, red[wi][r + half * 8]);
      rm[r] = x;
    }
    __syncthreads();

    // exp (masked entries underflow to 0) + row sum
    float ls[8];
    #pragma unroll
    for (int r = 0; r < 8; ++r) ls[r] = 0.f;
    #pragma unroll
    for (int t = 0; t < 8; ++t) {
      #pragma unroll
      for (int r = 0; r < 8; ++r) {
        const float p = __expf(W[t][r] - rm[r]);
        W[t][r] = p;
        ls[r] += p;
      }
    }
    #pragma unroll
    for (int r = 0; r < 8; ++r) {
      float x = ls[r];
      x += __shfl_xor(x, 1, 32);
      x += __shfl_xor(x, 2, 32);
      x += __shfl_xor(x, 4, 32);
      x += __shfl_xor(x, 8, 32);
      ls[r] = x;
    }
    if (rl == 0) {
      #pragma unroll
      for (int r = 0; r < 8; ++r) red[wv][r + half * 8] = ls[r];
    }
    __syncthreads();
    if (threadIdx.x < 16) {
      float s = 0.f;
      #pragma unroll
      for (int wi = 0; wi < 8; ++wi) s += red[wi][threadIdx.x];
      rs[((size_t)v * B + b) * P + pt * 16 + threadIdx.x] = s;
    }

    // store unnormalized probs as f16 A-fragments (LDS tile transpose)
    #pragma unroll
    for (int t = 0; t < 8; ++t) {
      #pragma unroll
      for (int r = 0; r < 8; ++r) ldsT[wv][r + half * 8][rl] = (_Float16)W[t][r];
      __builtin_amdgcn_wave_barrier();
      alignas(16) _Float16 ov[8];
      #pragma unroll
      for (int e = 0; e < 8; ++e) ov[e] = ldsT[wv][rl][half * 8 + e];
      const int mtile = wv * 8 + t;
      const size_t base = (((size_t)(v * B + b) * PT + pt) * KM + (mtile >> 1)) * 512
                          + lane * 16 + (mtile & 1) * 8;
      *(v8h*)(pf + base) = *(const v8h*)ov;
      __builtin_amdgcn_wave_barrier();
    }
    __syncthreads();
  }
}

// ---------------------------------------------------------------------------
// Kernel 5: out[v] = (probs[v] @ mention) / rowsum ; v=0 -> h, v=1 -> t
// ---------------------------------------------------------------------------
__global__ __launch_bounds__(256) void pv_kernel(
    const _Float16* __restrict__ pf, const _Float16* __restrict__ mf,
    const float* __restrict__ rs, float* __restrict__ out) {
  const int lane = threadIdx.x & 31, wv = threadIdx.x >> 5;
  const int half = lane >> 4, rl = lane & 15;
  int gid = blockIdx.x * 8 + wv;
  const int ht = gid % HT; gid /= HT;
  const int pt = gid % PT; gid /= PT;
  const int b  = gid % B;
  const int v  = gid / B;

  const size_t abase = (((size_t)(v * B + b) * PT + pt) * KM) * 512 + lane * 16;
  const size_t bbase = ((size_t)(b * HT + ht) * KM) * 512 + lane * 16;
  v8f acc = {};
  for (int m32 = 0; m32 < KM; ++m32) {
    __builtin_prefetch(pf + abase + (size_t)(m32 + 4) * 512, 0, 1);
    __builtin_prefetch(mf + bbase + (size_t)(m32 + 4) * 512, 0, 1);
    v16h a  = *(const v16h*)(pf + abase + (size_t)m32 * 512);
    v16h bb = *(const v16h*)(mf + bbase + (size_t)m32 * 512);
    acc = __builtin_amdgcn_wmma_f32_16x16x32_f16(false, a, false, bb,
                                                 (short)0, acc, false, false);
  }
  #pragma unroll
  for (int r = 0; r < 8; ++r) {
    const int prow = pt * 16 + r + half * 8;
    const float s = rs[((size_t)v * B + b) * P + prow];
    out[(((size_t)v * B + b) * P + prow) * H + ht * 16 + rl] = acc[r] / s;
  }
}

// ---------------------------------------------------------------------------
extern "C" void kernel_launch(void* const* d_in, const int* in_sizes, int n_in,
                              void* d_out, int out_size, void* d_ws, size_t ws_size,
                              hipStream_t stream) {
  const float* res     = (const float*)d_in[0];
  const float* mention = (const float*)d_in[1];
  const float* h_mask  = (const float*)d_in[2];
  const float* t_mask  = (const float*)d_in[3];
  const float* q_w     = (const float*)d_in[4];
  const float* q_b     = (const float*)d_in[5];
  const float* k_w     = (const float*)d_in[6];
  const float* k_b     = (const float*)d_in[7];
  float* out = (float*)d_out;

  char* ws = (char*)d_ws;
  size_t off = 0;
  _Float16* qf = (_Float16*)(ws + off); off += (size_t)B * P * H * 2;      // 64 MB
  _Float16* kf = (_Float16*)(ws + off); off += (size_t)B * M * H * 2;      //  8 MB
  _Float16* mf = (_Float16*)(ws + off); off += (size_t)B * M * H * 2;      //  8 MB
  _Float16* pf = (_Float16*)(ws + off); off += (size_t)2 * B * P * M * 2;  // 134 MB
  float*    rs = (float*)(ws + off);    off += (size_t)2 * B * P * 4;      // 256 KB

  qproj_kernel<<<(B * PT * HT) / 4, 128, 0, stream>>>(res, q_w, q_b, qf);
  kproj_kernel<<<(B * MT * HT) / 4, 128, 0, stream>>>(mention, k_w, k_b, kf);
  mconv_kernel<<<(B * HT * KM) / 4, 128, 0, stream>>>(mention, mf);
  attn_softmax_kernel<<<B * PT, 256, 0, stream>>>(qf, kf, h_mask, t_mask, pf, rs);
  pv_kernel<<<(2 * B * PT * HT) / 8, 256, 0, stream>>>(pf, mf, rs, out);
}
